// TanhBasisActivationLayerAffine_79748952752384
// MI455X (gfx1250) — compile-verified, matrix operand order checked by reading, not run on hardware
//
#include <hip/hip_runtime.h>

#define BATCH     4096
#define IN_COUNT  1024
#define OUT_COUNT 1024
#define NUM_BASIS 8
#define KDIM      (IN_COUNT * NUM_BASIS)   // 8192

typedef __attribute__((ext_vector_type(16))) _Float16 v16h;
typedef __attribute__((ext_vector_type(8)))  float    v8f;

union FragU  { v16h v; uint4 q[2]; };
union Half8U { _Float16 h[8]; uint4 q; };

// ---------------------------------------------------------------------------
// Kernel 1: basis[b, i*8+m] = sigmoid(2*slope*(alpha[i]*x[b,i]+beta[i] - center))
// (0.5*(1+tanh(z)) == sigmoid(2z)); output fp16, 16B packed store per thread.
// ---------------------------------------------------------------------------
__global__ __launch_bounds__(256) void tba_basis_kernel(
    const float* __restrict__ x,
    const float* __restrict__ centers,   // (IN_COUNT, 8)
    const float* __restrict__ slopes,    // (IN_COUNT, 8)
    const float* __restrict__ alpha,     // (IN_COUNT)
    const float* __restrict__ beta,      // (IN_COUNT)
    _Float16* __restrict__ basisH)       // (BATCH, KDIM) fp16
{
    int idx = blockIdx.x * blockDim.x + threadIdx.x;   // over BATCH*IN_COUNT
    if (idx >= BATCH * IN_COUNT) return;
    int i = idx & (IN_COUNT - 1);

    float u = alpha[i] * x[idx] + beta[i];

    Half8U out;
#pragma unroll
    for (int m = 0; m < NUM_BASIS; ++m) {
        float z = slopes[i * NUM_BASIS + m] * (u - centers[i * NUM_BASIS + m]);
        float s = 1.0f / (1.0f + __expf(-2.0f * z));   // == 0.5*(1+tanh(z))
        out.h[m] = (_Float16)s;
    }
    *reinterpret_cast<uint4*>(basisH + (size_t)idx * NUM_BASIS) = out.q;
}

// ---------------------------------------------------------------------------
// Kernel 2: coeffs f32 -> f16, 8 elements per thread (2x float4 load, 1x uint4 store)
// ---------------------------------------------------------------------------
__global__ __launch_bounds__(256) void tba_cvt_kernel(
    const float* __restrict__ c, _Float16* __restrict__ ch, int n8)
{
    int idx = blockIdx.x * blockDim.x + threadIdx.x;
    if (idx >= n8) return;
    const float4* p = reinterpret_cast<const float4*>(c) + (size_t)idx * 2;
    float4 a = p[0], b = p[1];
    Half8U o;
    o.h[0] = (_Float16)a.x; o.h[1] = (_Float16)a.y;
    o.h[2] = (_Float16)a.z; o.h[3] = (_Float16)a.w;
    o.h[4] = (_Float16)b.x; o.h[5] = (_Float16)b.y;
    o.h[6] = (_Float16)b.z; o.h[7] = (_Float16)b.w;
    *(reinterpret_cast<uint4*>(ch) + idx) = o.q;
}

// ---------------------------------------------------------------------------
// Kernel 3: Y(4096x1024) = basisH(4096x8192) * coeffsH(1024x8192)^T
// Both operands K-major. Per-lane fragment layout for 16-bit WMMA on wave32:
//   lane<16 : row = lane,    K chunks {0..7} and {16..23}
//   lane>=16: row = lane-16, K chunks {8..15} and {24..31}
// -> two global_load_b128 per fragment. Wave tile 64x64 = 4x4 v_wmma per K=32.
// Block = 4 waves (2x2) -> 128x128 tile. Grid = (4096/128, 1024/128) = (32, 8).
// ---------------------------------------------------------------------------
__global__ __launch_bounds__(128) void tba_gemm_kernel(
    const _Float16* __restrict__ A,   // (BATCH, KDIM)
    const _Float16* __restrict__ B,   // (OUT_COUNT, KDIM)
    float* __restrict__ C)            // (BATCH, OUT_COUNT)
{
    const int lane = threadIdx.x & 31;
    const int wave = threadIdx.x >> 5;

    const int wm = (wave >> 1) * 64;          // wave row offset in block
    const int wn = (wave & 1) * 64;           // wave col offset in block
    const int bm = blockIdx.x * 128;
    const int bn = blockIdx.y * 128;

    const int row16 = lane & 15;
    const int khalf = (lane >> 4) * 8;        // 0 or 8 halves

    const _Float16* aBase = A + (size_t)(bm + wm + row16) * KDIM + khalf;
    const _Float16* bBase = B + (size_t)(bn + wn + row16) * KDIM + khalf;

    v8f acc[4][4];
#pragma unroll
    for (int i = 0; i < 4; ++i)
#pragma unroll
        for (int j = 0; j < 4; ++j)
            acc[i][j] = (v8f)(0.0f);

    for (int k = 0; k < KDIM; k += 32) {
        FragU af[4], bf[4];
#pragma unroll
        for (int t = 0; t < 4; ++t) {
            const uint4* pa = reinterpret_cast<const uint4*>(aBase + (size_t)(t * 16) * KDIM + k);
            af[t].q[0] = pa[0];               // K chunk at +0
            af[t].q[1] = pa[2];               // K chunk at +16 halves (=32B)
            const uint4* pb = reinterpret_cast<const uint4*>(bBase + (size_t)(t * 16) * KDIM + k);
            bf[t].q[0] = pb[0];
            bf[t].q[1] = pb[2];
        }
#pragma unroll
        for (int i = 0; i < 4; ++i)
#pragma unroll
            for (int j = 0; j < 4; ++j)
                acc[i][j] = __builtin_amdgcn_wmma_f32_16x16x32_f16(
                    /*neg_a=*/false, af[i].v, /*neg_b=*/false, bf[j].v,
                    /*c_mod=*/(short)0, acc[i][j],
                    /*reuse_a=*/false, /*reuse_b=*/false);
    }

    // D layout: VGPR r -> row r (lanes 0-15) or r+8 (lanes 16-31); col = lane&15
    const int colIdx = lane & 15;
    const int rowOff = (lane >> 4) * 8;
#pragma unroll
    for (int i = 0; i < 4; ++i) {
#pragma unroll
        for (int j = 0; j < 4; ++j) {
            float* cPtr = C + (size_t)(bm + wm + i * 16 + rowOff) * OUT_COUNT
                            + (bn + wn + j * 16 + colIdx);
#pragma unroll
            for (int r = 0; r < 8; ++r)
                cPtr[(size_t)r * OUT_COUNT] = acc[i][j][r];
        }
    }
}

// ---------------------------------------------------------------------------
extern "C" void kernel_launch(void* const* d_in, const int* in_sizes, int n_in,
                              void* d_out, int out_size, void* d_ws, size_t ws_size,
                              hipStream_t stream)
{
    const float* x       = (const float*)d_in[0];   // (4096,1024)
    const float* coeffs  = (const float*)d_in[1];   // (1024,1024,8)
    const float* centers = (const float*)d_in[2];   // (1024,8)
    const float* slopes  = (const float*)d_in[3];   // (1024,8)
    const float* alpha   = (const float*)d_in[4];   // (1024)
    const float* beta    = (const float*)d_in[5];   // (1024)
    float*       y       = (float*)d_out;           // (4096,1024)

    // Workspace layout: basisH (64 MB fp16) | coeffsH (16 MB fp16)
    _Float16* basisH  = (_Float16*)d_ws;
    _Float16* coeffsH = (_Float16*)((char*)d_ws + (size_t)BATCH * KDIM * sizeof(_Float16));

    {   // basis expansion
        int n = BATCH * IN_COUNT;
        tba_basis_kernel<<<(n + 255) / 256, 256, 0, stream>>>(
            x, centers, slopes, alpha, beta, basisH);
    }
    {   // coeffs f32 -> f16
        int n8 = (OUT_COUNT * KDIM) / 8;
        tba_cvt_kernel<<<(n8 + 255) / 256, 256, 0, stream>>>(coeffs, coeffsH, n8);
    }
    {   // WMMA GEMM
        dim3 grid(BATCH / 128, OUT_COUNT / 128);
        tba_gemm_kernel<<<grid, 128, 0, stream>>>(basisH, coeffsH, y);
    }
}
